// WeightDemodulation_39290360824780
// MI455X (gfx1250) — compile-verified, matrix operand order checked by reading, not run on hardware
//
#include <hip/hip_runtime.h>
#include <hip/hip_bf16.h>
#include <math.h>

typedef __bf16 bf16;
typedef __attribute__((ext_vector_type(16))) __bf16 v16bf;
typedef __attribute__((ext_vector_type(8)))  float  v8f;
typedef unsigned int __attribute__((ext_vector_type(4))) u32x4;
typedef int          __attribute__((ext_vector_type(4))) i32x4;
typedef int          __attribute__((ext_vector_type(8))) i32x8;

#define CIN   512
#define COUT  512
#define HH    64
#define WWID  64
#define HW    4096
#define NTAP  9
#define KTOT  (CIN * NTAP)   // 4608
#define AP    40             // padded LDS row stride in bf16 (80B: 16B aligned, 20-bank stride)
#define HROW  66             // halo width  (64 + 2)
#define HCOL  4              // halo height (2 + 2)

#if __has_builtin(__builtin_amdgcn_tensor_load_to_lds) && __has_builtin(__builtin_amdgcn_s_wait_tensorcnt)
#define USE_TDM 1
#else
#define USE_TDM 0
#endif

// ---------------------------------------------------------------------------
// Stage 1: lr-equalize + modulate + demodulate weights, emit bf16
// A layout: [b][tap][co][ci]
// ---------------------------------------------------------------------------
__global__ __launch_bounds__(256)
void prep_weights(const float* __restrict__ w, const float* __restrict__ cond,
                  bf16* __restrict__ A)
{
    __shared__ float mval[KTOT];
    __shared__ float red[256];
    const int co = blockIdx.x;
    const int b  = blockIdx.y;
    const float scale = 0.02083333333333f;   // sqrt(GAIN/fan_in) = sqrt(2/4608) = 1/48
    const float* wr = w + (size_t)co * KTOT;
    const float* cb = cond + (size_t)b * CIN;

    float s = 0.f;
    for (int idx = threadIdx.x; idx < KTOT; idx += 256) {
        const int ci = idx / 9;
        const float v = wr[idx] * scale * cb[ci];
        mval[idx] = v;
        s += v * v;
    }
    red[threadIdx.x] = s;
    __syncthreads();
    for (int off = 128; off > 0; off >>= 1) {
        if (threadIdx.x < off) red[threadIdx.x] += red[threadIdx.x + off];
        __syncthreads();
    }
    const float inv = 1.0f / sqrtf(red[0] + 1e-6f);

    for (int idx = threadIdx.x; idx < KTOT; idx += 256) {
        const int ci = idx / 9;
        const int t  = idx - ci * 9;
        A[((size_t)(b * NTAP + t) * COUT + co) * CIN + ci] = (bf16)(mval[idx] * inv);
    }
}

#if USE_TDM
// ---------------------------------------------------------------------------
// TDM descriptor (ISA 8.3/8.4): 2D tile 32(ci) x 128(co) bf16 rows of 64B,
// tensor_dim0_stride=512, LDS pad 4 DWORDs every 16 DWORDs -> 80B row stride.
// ---------------------------------------------------------------------------
__device__ __forceinline__ void tdm_load_A(const bf16* gsrc, unsigned lds_off)
{
    const unsigned long long ga = (unsigned long long)(const void*)gsrc;
    u32x4 g0;
    g0.x = 1u;                                   // count=1, user descriptor
    g0.y = lds_off;                              // lds_addr (bytes)
    g0.z = (unsigned)(ga & 0xffffffffu);         // global_addr[31:0]
    g0.w = (unsigned)((ga >> 32) & 0x01ffffffu)  // global_addr[56:32]
         | (2u << 30);                           // type = 2 ("image")
    i32x8 g1;
    g1[0] = (1 << 16)          // data_size = 2 bytes
          | (1 << 20)          // pad_enable
          | (3 << 22)          // pad_interval: 16 DWORDs (one 64B tile row)
          | (3 << 25);         // pad_amount:   4 DWORDs -> 80B LDS row stride
    g1[1] = (CIN & 0xffff) << 16;                    // tensor_dim0[15:0]
    g1[2] = ((CIN >> 16) & 0xffff) | ((COUT & 0xffff) << 16); // dim0 hi | dim1 lo
    g1[3] = ((COUT >> 16) & 0xffff) | (32 << 16);    // dim1 hi | tile_dim0 = 32
    g1[4] = 128;                                     // tile_dim1 = 128, tile_dim2 = 0
    g1[5] = CIN;                                     // tensor_dim0_stride[31:0]
    g1[6] = 0;
    g1[7] = 0;
    const i32x4 gz = {0, 0, 0, 0};
#if __clang_major__ >= 23
    const i32x8 gz8 = {0, 0, 0, 0, 0, 0, 0, 0};
    __builtin_amdgcn_tensor_load_to_lds(g0, g1, gz, gz, gz8, 0);
#else
    __builtin_amdgcn_tensor_load_to_lds(g0, g1, gz, gz, 0);
#endif
}
#endif

// ---------------------------------------------------------------------------
// Stage 2: per-batch implicit GEMM conv, bf16 WMMA.
// grid (COUT/128, HW/128, B); block 256 = 8 wave32.
// Loop: c outer (16 Cin chunks) stages one 32ci x 4row x 66col halo slab,
//       t inner (9 taps) reads shifted windows; A tiles double-buffered
//       (TDM DMA when available), one barrier per tap.
// ---------------------------------------------------------------------------
__global__ __launch_bounds__(256, 2)
void conv_bf16_wmma(const float* __restrict__ x, const bf16* __restrict__ A,
                    const float* __restrict__ bias, float* __restrict__ out)
{
    __shared__ bf16 Abuf[2][128 * AP];          // [co][ci] weight tiles (double buffered)
    __shared__ bf16 Bh[HCOL * HROW * AP];       // [halo row*66+col][ci]

    const int tid  = threadIdx.x;
    const int b    = blockIdx.z;
    const int co0  = blockIdx.x * 128;
    const int p0   = blockIdx.y * 128;
    const int h0   = p0 >> 6;

    const int lane = tid & 31;
    const int wv   = tid >> 5;
    const int wm   = wv & 3;                  // 4 co subtiles of 32
    const int wn   = wv >> 2;                 // 2 px subtiles of 64 (= image row wn)
    const int rsel = lane & 15;
    const int koff = (lane & 16) ? 8 : 0;     // K chunk select per ISA 16-bit layout

    const int arow = tid >> 1;                // manual A loader: co row
    const int acol = (tid & 1) << 4;          // ci 0 / 16

    const float* xb = x + (size_t)b * CIN * HW;
    const bf16*  Ab = A + (size_t)b * NTAP * COUT * CIN;

    v8f acc[2][4];
    const v8f vz = {0.f, 0.f, 0.f, 0.f, 0.f, 0.f, 0.f, 0.f};
#pragma unroll
    for (int i = 0; i < 2; ++i)
#pragma unroll
        for (int j = 0; j < 4; ++j) acc[i][j] = vz;

    // ---- prologue: stage A tile kk=0 into buffer 0 ----
#if USE_TDM
    if (wv == 0) tdm_load_A(Ab + (size_t)co0 * CIN, (unsigned)(size_t)&Abuf[0][0]);
#else
    {
        const uint4* ag = (const uint4*)(Ab + (size_t)(co0 + arow) * CIN + acol);
        const uint4 a0 = ag[0];
        const uint4 a1 = ag[1];
        uint4* as = (uint4*)&Abuf[0][arow * AP + acol];
        as[0] = a0; as[1] = a1;
    }
#endif

    for (int c = 0; c < 16; ++c) {
        // ---- stage halo slab: 32 ci x 4 rows x 66 cols, f32 -> bf16, transposed ----
        const float* xc = xb + (size_t)c * 32 * HW;
        for (int k = 0; k < 33; ++k) {
            const int j   = k * 256 + tid;
            const int ci  = j / (HCOL * HROW);
            const int rem = j - ci * (HCOL * HROW);
            const int row = rem / HROW;
            const int col = rem - row * HROW;
            const int hh  = h0 - 1 + row;
            const int wp  = col - 1;
            const int hc  = hh < 0 ? 0 : (hh > 63 ? 63 : hh);
            const int wc  = wp < 0 ? 0 : (wp > 63 ? 63 : wp);
            const float f = xc[((size_t)ci * HH + hc) * WWID + wc];
            const bool  v = (hh == hc) & (wp == wc);
            Bh[(row * HROW + col) * AP + ci] = (bf16)(v ? f : 0.0f);
        }
#if USE_TDM
        if (wv == 0) __builtin_amdgcn_s_wait_tensorcnt(0);
#endif
        __syncthreads();   // halo + current A buffer visible

        for (int t = 0; t < NTAP; ++t) {
            const int kk = c * NTAP + t;
            const bool pf = (kk < 16 * NTAP - 1);
            const int nk  = pf ? kk + 1 : kk;
            const int ncc = nk / NTAP;
            const int nt  = nk - ncc * NTAP;

            // ---- prefetch next A tile into the other buffer ----
#if USE_TDM
            if (pf && wv == 0)
                tdm_load_A(Ab + ((size_t)nt * COUT + co0) * CIN + (size_t)ncc * 32,
                           (unsigned)(size_t)&Abuf[(kk + 1) & 1][0]);
#else
            uint4 a0, a1;
            if (pf) {
                const uint4* ag = (const uint4*)
                    (Ab + ((size_t)nt * COUT + co0 + arow) * CIN + (size_t)ncc * 32 + acol);
                a0 = ag[0];
                a1 = ag[1];
            }
#endif
            // ---- fragments ----
            const bf16* Acur = &Abuf[kk & 1][0];
            const int dh = t / 3 - 1;
            const int dw = t - (t / 3) * 3 - 1;
            const int hbase = (wn + dh + 1) * HROW + (dw + 1) + rsel;

            v16bf af[2], bfrag[4];
#pragma unroll
            for (int i = 0; i < 2; ++i) {
                const int r = wm * 32 + i * 16 + rsel;
                union { v16bf v; uint4 q[2]; } u;
                u.q[0] = *(const uint4*)&Acur[r * AP + koff];
                u.q[1] = *(const uint4*)&Acur[r * AP + koff + 16];
                af[i] = u.v;
            }
#pragma unroll
            for (int j = 0; j < 4; ++j) {
                const int pr = hbase + j * 16;
                union { v16bf v; uint4 q[2]; } u;
                u.q[0] = *(const uint4*)&Bh[pr * AP + koff];
                u.q[1] = *(const uint4*)&Bh[pr * AP + koff + 16];
                bfrag[j] = u.v;
            }

#pragma unroll
            for (int i = 0; i < 2; ++i)
#pragma unroll
                for (int j = 0; j < 4; ++j)
                    acc[i][j] = __builtin_amdgcn_wmma_f32_16x16x32_bf16(
                        false, af[i], false, bfrag[j], (short)0, acc[i][j],
                        false, false);

#if USE_TDM
            if (wv == 0) __builtin_amdgcn_s_wait_tensorcnt(0);
#else
            if (pf) {
                uint4* as = (uint4*)&Abuf[(kk + 1) & 1][arow * AP + acol];
                as[0] = a0; as[1] = a1;
            }
#endif
            __syncthreads();
        }
    }

    // ---- epilogue: D layout: VGPR r -> M = r + (lane>=16 ? 8:0), N = lane&15 ----
    const int mhi = (lane & 16) ? 8 : 0;
    float* ob = out + (size_t)b * COUT * HW;
#pragma unroll
    for (int i = 0; i < 2; ++i) {
#pragma unroll
        for (int r = 0; r < 8; ++r) {
            const int co = co0 + wm * 32 + i * 16 + mhi + r;
            const float bv = bias[co];
#pragma unroll
            for (int j = 0; j < 4; ++j) {
                const int p = p0 + wn * 64 + j * 16 + rsel;
                ob[(size_t)co * HW + p] = acc[i][j][r] + bv;
            }
        }
    }
}

// ---------------------------------------------------------------------------
extern "C" void kernel_launch(void* const* d_in, const int* in_sizes, int n_in,
                              void* d_out, int out_size, void* d_ws, size_t ws_size,
                              hipStream_t stream)
{
    const float* inputs = (const float*)d_in[0];   // [16,512,64,64]
    const float* cond   = (const float*)d_in[1];   // [16,512]
    const float* weight = (const float*)d_in[2];   // [512,512,3,3]
    const float* bias   = (const float*)d_in[3];   // [512]
    float* out = (float*)d_out;                    // [16,512,64,64]

    bf16* A = (bf16*)d_ws;  // [16][9][512][512] bf16 = 75.5 MB modulated weights

    prep_weights<<<dim3(COUT, 16), 256, 0, stream>>>(weight, cond, A);
    conv_bf16_wmma<<<dim3(COUT / 128, HW / 128, 16), 256, 0, stream>>>(inputs, A, bias, out);
}
